// MyGAT_88802743812336
// MI455X (gfx1250) — compile-verified, matrix operand order checked by reading, not run on hardware
//
#include <hip/hip_runtime.h>
#include <hip/hip_fp16.h>
#include <math.h>

typedef __attribute__((ext_vector_type(16))) _Float16 v16h;
typedef __attribute__((ext_vector_type(8)))  float    v8f;

#define NN 100000L
#define EE 1600000L

// ---------------------------------------------------------------- utilities

__device__ __forceinline__ float warp_sum16(float v) {
  // sum across the 16-lane group (lanes sharing bit4) of a wave32
  v += __shfl_xor(v, 1, 32);
  v += __shfl_xor(v, 2, 32);
  v += __shfl_xor(v, 4, 32);
  v += __shfl_xor(v, 8, 32);
  return v;
}

__device__ __forceinline__ void atomicMaxF(float* addr, float val) {
  unsigned int* ua = (unsigned int*)addr;
  unsigned int old = *ua;
  while (__uint_as_float(old) < val) {
    unsigned int assumed = old;
    old = atomicCAS(ua, assumed, __float_as_uint(val));
    if (old == assumed) break;
  }
}

// load 8 contiguous f32 (32B aligned) -> av[0..7]
__device__ __forceinline__ void load8(const float* p, float* av) {
  float4 a = *(const float4*)(p);
  float4 b = *(const float4*)(p + 4);
  av[0]=a.x; av[1]=a.y; av[2]=a.z; av[3]=a.w;
  av[4]=b.x; av[5]=b.y; av[6]=b.z; av[7]=b.w;
}

// ------------------------------------------------------- node-side WMMA GEMM
// D[M,64] = op(A[M,K]) @ W[K,64]  (op = optional relu on A)
// optional fused epilogue: a_src[m,h] = sum_c D[m,16h+c]*att_s[h,c] (same a_dst)
// optional fused bias add (output head).
template<int K, int RELU_A, int DO_ATT, int BIAS>
__global__ void gemm_node_kernel(const float* __restrict__ A,
                                 const float* __restrict__ W,
                                 const float* __restrict__ att_s,
                                 const float* __restrict__ att_d,
                                 const float* __restrict__ bias,
                                 float* __restrict__ D,
                                 float* __restrict__ a_src,
                                 float* __restrict__ a_dst)
{
  const int lane = threadIdx.x & 31;
  const int w    = threadIdx.x >> 5;      // 0..3 : head == N-tile
  const int half = lane >> 4;
  const int l16  = lane & 15;
  const long long row0 = (long long)blockIdx.x * 16;
  const int nc0  = w * 16;
  const int col  = nc0 + l16;

  // preload B fragments (W is tiny: L2/L0 resident)
  v16h bfrag[K / 32];
#pragma unroll
  for (int ks = 0; ks < K / 32; ++ks) {
    const int k0 = ks * 32;
#pragma unroll
    for (int j = 0; j < 8; ++j) {
      bfrag[ks][j]     = (_Float16)W[(k0 + half * 8 + j) * 64 + col];
      bfrag[ks][j + 8] = (_Float16)W[(k0 + 16 + half * 8 + j) * 64 + col];
    }
  }

  const float* Arow = A + (row0 + l16) * K;
  v8f acc = {};
#pragma unroll
  for (int ks = 0; ks < K / 32; ++ks) {
    const int k0 = ks * 32;
    float av[16];
    load8(Arow + k0 + half * 8,      av);
    load8(Arow + k0 + 16 + half * 8, av + 8);
    v16h afrag;
#pragma unroll
    for (int j = 0; j < 16; ++j) {
      float t = av[j];
      if (RELU_A) t = fmaxf(t, 0.f);
      afrag[j] = (_Float16)t;
    }
    acc = __builtin_amdgcn_wmma_f32_16x16x32_f16(
        false, afrag, false, bfrag[ks], (short)0, acc, false, false);
  }

  // store D tile
#pragma unroll
  for (int r = 0; r < 8; ++r) {
    const long long m = row0 + half * 8 + r;
    float dv = acc[r];
    if (BIAS) dv += bias[col];
    D[m * 64 + col] = dv;
  }

  if (DO_ATT) {
    const float as = att_s[nc0 + l16];
    const float ad = att_d[nc0 + l16];
#pragma unroll
    for (int r = 0; r < 8; ++r) {
      const float ss = warp_sum16(acc[r] * as);
      const float sd = warp_sum16(acc[r] * ad);
      if (l16 == r) {
        const long long m = row0 + half * 8 + r;
        a_src[m * 4 + w] = ss;
        a_dst[m * 4 + w] = sd;
      }
    }
  }
}

// ------------------------------------------------------- edge-side WMMA GEMM
// MEAN3: A = mean over 3 of edge_attr[E,3,64] (fused, edge_attr read once)
// else : A = e1[E,64]
// Always fuses a_e[e,h] = sum_c D[e,16h+c]*att_e[h,c]; optionally stores D(e1).
template<int MEAN3, int STORE_D>
__global__ void gemm_edge_kernel(const float* __restrict__ A,
                                 const float* __restrict__ W,
                                 const float* __restrict__ att_e,
                                 float* __restrict__ Dout,
                                 float* __restrict__ a_e)
{
  const int lane = threadIdx.x & 31;
  const int w    = threadIdx.x >> 5;
  const int half = lane >> 4;
  const int l16  = lane & 15;
  const long long row0 = (long long)blockIdx.x * 16;
  const int nc0  = w * 16;
  const int col  = nc0 + l16;

  v16h bfrag[2];
#pragma unroll
  for (int ks = 0; ks < 2; ++ks) {
    const int k0 = ks * 32;
#pragma unroll
    for (int j = 0; j < 8; ++j) {
      bfrag[ks][j]     = (_Float16)W[(k0 + half * 8 + j) * 64 + col];
      bfrag[ks][j + 8] = (_Float16)W[(k0 + 16 + half * 8 + j) * 64 + col];
    }
  }

  const float* Arow = A + (row0 + l16) * (MEAN3 ? 192 : 64);
  v8f acc = {};
#pragma unroll
  for (int ks = 0; ks < 2; ++ks) {
    const int k0 = ks * 32;
    float av[16];
    if (MEAN3) {
      float t0[8], t1[8], t2[8];
#pragma unroll
      for (int g = 0; g < 2; ++g) {
        const int base = k0 + 16 * g + half * 8;
        load8(Arow + base,       t0);
        load8(Arow + base + 64,  t1);
        load8(Arow + base + 128, t2);
#pragma unroll
        for (int j = 0; j < 8; ++j)
          av[8 * g + j] = (t0[j] + t1[j] + t2[j]) * (1.f / 3.f);
      }
    } else {
      load8(Arow + k0 + half * 8,      av);
      load8(Arow + k0 + 16 + half * 8, av + 8);
    }
    v16h afrag;
#pragma unroll
    for (int j = 0; j < 16; ++j) afrag[j] = (_Float16)av[j];
    acc = __builtin_amdgcn_wmma_f32_16x16x32_f16(
        false, afrag, false, bfrag[ks], (short)0, acc, false, false);
  }

  if (STORE_D) {
#pragma unroll
    for (int r = 0; r < 8; ++r)
      Dout[(row0 + half * 8 + r) * 64 + col] = acc[r];
  }

  const float ae = att_e[nc0 + l16];
#pragma unroll
  for (int r = 0; r < 8; ++r) {
    const float s = warp_sum16(acc[r] * ae);
    if (l16 == r) a_e[(row0 + half * 8 + r) * 4 + w] = s;
  }
}

// ------------------------------------------------------------- scatter stage

__global__ void init_kernel(float* __restrict__ hbuf, float* __restrict__ m,
                            float* __restrict__ denom) {
  const long long gid = (long long)blockIdx.x * blockDim.x + threadIdx.x;
  if (gid < NN * 64) hbuf[gid] = 0.f;
  if (gid < NN * 4) { m[gid] = -INFINITY; denom[gid] = 0.f; }
}

__global__ void logits_kernel(const long long* __restrict__ ei,
                              const float* __restrict__ a_src,
                              const float* __restrict__ a_dst,
                              const float* __restrict__ a_e,
                              float* __restrict__ elog,
                              float* __restrict__ m)
{
  const long long gid = (long long)blockIdx.x * blockDim.x + threadIdx.x;
  if (gid >= EE * 4) return;
  const long long e = gid >> 2; const int hh = gid & 3;
  const long long s = ei[e];
  const long long d = ei[EE + e];
  float a = a_src[s * 4 + hh] + a_dst[d * 4 + hh] + a_e[gid];
  a = (a > 0.f) ? a : 0.2f * a;          // leaky_relu(0.2)
  elog[gid] = a;
  atomicMaxF(&m[d * 4 + hh], a);
}

__global__ void expsum_kernel(const long long* __restrict__ ei,
                              float* __restrict__ elog,   // in: logit, out: exp
                              const float* __restrict__ m,
                              float* __restrict__ denom)
{
  const long long gid = (long long)blockIdx.x * blockDim.x + threadIdx.x;
  if (gid >= EE * 4) return;
  const long long e = gid >> 2; const int hh = gid & 3;
  const long long d = ei[EE + e];
  float mm = m[d * 4 + hh];
  if (mm < -3e38f) mm = 0.f;             // isfinite guard
  const float v = expf(elog[gid] - mm);
  elog[gid] = v;
  atomicAdd(&denom[d * 4 + hh], v);
}

__global__ void aggregate_kernel(const long long* __restrict__ ei,
                                 const float* __restrict__ alpha,
                                 const float* __restrict__ denom,
                                 const float* __restrict__ xp,
                                 float* __restrict__ hout)
{
  const long long gid = (long long)blockIdx.x * blockDim.x + threadIdx.x;
  if (gid >= EE * 4) return;
  const long long e = gid >> 2; const int hh = gid & 3;
  const long long s = ei[e];
  const long long d = ei[EE + e];
  const float w = alpha[gid] / fmaxf(denom[d * 4 + hh], 1e-16f);
  const float* xs = xp + s * 64 + hh * 16;
  float* ho = hout + d * 64 + hh * 16;
  float4 v0 = *(const float4*)(xs);
  float4 v1 = *(const float4*)(xs + 4);
  float4 v2 = *(const float4*)(xs + 8);
  float4 v3 = *(const float4*)(xs + 12);
  const float vv[16] = {v0.x,v0.y,v0.z,v0.w, v1.x,v1.y,v1.z,v1.w,
                        v2.x,v2.y,v2.z,v2.w, v3.x,v3.y,v3.z,v3.w};
#pragma unroll
  for (int j = 0; j < 16; ++j) atomicAdd(&ho[j], w * vv[j]);
}

// ------------------------------------------------------------------- driver

extern "C" void kernel_launch(void* const* d_in, const int* in_sizes, int n_in,
                              void* d_out, int out_size, void* d_ws, size_t ws_size,
                              hipStream_t stream)
{
  (void)in_sizes; (void)n_in; (void)out_size; (void)ws_size;
  const float*     x         = (const float*)d_in[0];
  const long long* ei        = (const long long*)d_in[1];
  const float*     edge_attr = (const float*)d_in[2];
  const float* W1   = (const float*)d_in[3];
  const float* We1  = (const float*)d_in[4];
  const float* as1  = (const float*)d_in[5];
  const float* ad1  = (const float*)d_in[6];
  const float* ae1  = (const float*)d_in[7];
  const float* W2   = (const float*)d_in[8];
  const float* We2  = (const float*)d_in[9];
  const float* as2  = (const float*)d_in[10];
  const float* ad2  = (const float*)d_in[11];
  const float* ae2  = (const float*)d_in[12];
  const float* Wout = (const float*)d_in[13];
  const float* bout = (const float*)d_in[14];

  float* ws = (float*)d_ws;
  float* e1    = ws; ws += EE * 64;   // layer-1 edge features (layer-2 edge input)
  float* xp    = ws; ws += NN * 64;   // transformed node features
  float* hbuf  = ws; ws += NN * 64;   // aggregation target / next-layer input
  float* a_src = ws; ws += NN * 4;
  float* a_dst = ws; ws += NN * 4;
  float* mbuf  = ws; ws += NN * 4;
  float* denom = ws; ws += NN * 4;
  float* a_e   = ws; ws += EE * 4;
  float* elog  = ws; ws += EE * 4;

  const dim3 blk(128);
  const int nodeTiles = (int)(NN / 16);   // 6250
  const int edgeTiles = (int)(EE / 16);   // 100000
  const int initBlocks = (int)((NN * 64 + 255) / 256);
  const int edgeBlocks = (int)((EE * 4 + 255) / 256);

  // ---------------- layer 1 ----------------
  gemm_node_kernel<128, 0, 1, 0><<<nodeTiles, blk, 0, stream>>>(
      x, W1, as1, ad1, nullptr, xp, a_src, a_dst);
  gemm_edge_kernel<1, 1><<<edgeTiles, blk, 0, stream>>>(
      edge_attr, We1, ae1, e1, a_e);
  init_kernel<<<initBlocks, 256, 0, stream>>>(hbuf, mbuf, denom);
  logits_kernel<<<edgeBlocks, 256, 0, stream>>>(ei, a_src, a_dst, a_e, elog, mbuf);
  expsum_kernel<<<edgeBlocks, 256, 0, stream>>>(ei, elog, mbuf, denom);
  aggregate_kernel<<<edgeBlocks, 256, 0, stream>>>(ei, elog, denom, xp, hbuf);

  // ---------------- layer 2 ---------------- (relu fused into A-load)
  gemm_node_kernel<64, 1, 1, 0><<<nodeTiles, blk, 0, stream>>>(
      hbuf, W2, as2, ad2, nullptr, xp, a_src, a_dst);
  gemm_edge_kernel<0, 0><<<edgeTiles, blk, 0, stream>>>(
      e1, We2, ae2, nullptr, a_e);
  init_kernel<<<initBlocks, 256, 0, stream>>>(hbuf, mbuf, denom);
  logits_kernel<<<edgeBlocks, 256, 0, stream>>>(ei, a_src, a_dst, a_e, elog, mbuf);
  expsum_kernel<<<edgeBlocks, 256, 0, stream>>>(ei, elog, mbuf, denom);
  aggregate_kernel<<<edgeBlocks, 256, 0, stream>>>(ei, elog, denom, xp, hbuf);

  // ---------------- output head ------------ (relu fused, bias fused)
  gemm_node_kernel<64, 1, 0, 1><<<nodeTiles, blk, 0, stream>>>(
      hbuf, Wout, nullptr, nullptr, bout, (float*)d_out, nullptr, nullptr);
}
